// BayesianOddLayer_66443144069556
// MI455X (gfx1250) — matrix operand order repaired and checked by reading, not compiled, not run on hardware
//
#include <hip/hip_runtime.h>
#include <hip/hip_bf16.h>

typedef __attribute__((ext_vector_type(16))) _Float16 v16h;
typedef __attribute__((ext_vector_type(8)))  float    v8f;
typedef __attribute__((ext_vector_type(4)))  float    f32x4;

#define BATCH 16384
#define EDGES 2048
#define NVARS 512
#define KTOT  2560   // EDGES + NVARS
#define BK    128    // K-chunk staged through LDS (divides 2048 and 512)
#define NCHUNK (KTOT / BK)          // 20
#define XCHUNK (EDGES / BK)         // 16 chunks come from x, rest from llr
#define CLIP_TANH 10.0f

// ---------------------------------------------------------------------------
// Kernel 1: build the fused, gated weight matrix, transposed + f16.
//   WcT[n][k]  (n in [0,2048), k in [0,2560))
//     k <  2048 : z[k] * w_odd2even_mask[k][n] * odd_weights[k][n]
//     k >= 2048 : w_skipconn2even_mask[k-2048][n] * llr_weights[k-2048][n]
//   z[k] = (u[k] < sigmoid(dropout_logits[k]))
// ---------------------------------------------------------------------------
__global__ __launch_bounds__(256) void prep_weights(
    const float* __restrict__ u, const float* __restrict__ odd_w,
    const float* __restrict__ llr_w, const float* __restrict__ logits,
    const float* __restrict__ mask_o2e, const float* __restrict__ mask_skip,
    _Float16* __restrict__ wct) {
  size_t idx = (size_t)blockIdx.x * 256 + threadIdx.x;   // n * KTOT + k
  int n = (int)(idx / KTOT);
  int k = (int)(idx % KTOT);
  float w;
  if (k < EDGES) {
    float sig = 1.0f / (1.0f + __expf(-logits[k]));
    float z = (u[k] < sig) ? 1.0f : 0.0f;
    size_t o = (size_t)k * EDGES + n;
    w = mask_o2e[o] * odd_w[o] * z;
  } else {
    int v = k - EDGES;
    size_t o = (size_t)v * EDGES + n;
    w = mask_skip[o] * llr_w[o];
  }
  wct[idx] = (_Float16)w;
}

// ---------------------------------------------------------------------------
// Kernel 2: fused dual-GEMM + tanh epilogue.
// Block = 256 threads = 8 waves (wave32), arranged 4(M) x 2(N).
// Block tile 128M x 128N; wave tile 32M x 64N (8 f32 accumulators).
// B is staged per 128-K chunk into LDS in exact WMMA-operand order:
//   ldsB element group g = (s*8 + t)  (s = k-step 0..3, t = colgroup 0..7),
//   lane L reads 32 contiguous bytes at (g*32 + L)*32 -> conflict-free.
// WMMA operand k order used here: A lane half h holds K = h*16..+15
// for row M = lane&15; B lane half h holds K = h*16..+15 for col N = lane&15.
// ---------------------------------------------------------------------------
__global__ __launch_bounds__(256) void decoder_gemm(
    const float* __restrict__ x, const float* __restrict__ llr,
    const _Float16* __restrict__ wct, float* __restrict__ out) {
  __shared__ alignas(32) _Float16 ldsB[4 * 8 * 32 * 16];   // 32 KB

  const int tid  = threadIdx.x;
  const int lane = tid & 31;
  const int wave = tid >> 5;
  const int mw   = wave & 3;          // M-wave index (0..3)
  const int nw   = wave >> 2;         // N-wave index (0..1)
  const int h    = lane >> 4;         // lane half
  const int ml   = lane & 15;         // row (A) / col (B) within 16x16 tile

  const int m0   = blockIdx.y * 128 + mw * 32;    // wave's first row
  const int ncol0 = blockIdx.x * 128;             // block's first col

  // staging indices: this thread copies blocks rep*256+tid, rep = 0..3
  const int st_t  = (tid >> 5) & 7;   // colgroup 0..7
  const int st_L  = tid & 31;
  const int st_h  = st_L >> 4;
  const int st_m  = st_L & 15;
  const _Float16* __restrict__ wsrc =
      wct + (size_t)(ncol0 + st_t * 16 + st_m) * KTOT + st_h * 16;
  _Float16* __restrict__ ldst = &ldsB[(size_t)tid * 16];  // + rep*256*16

  // A row pointers (two 16-row strips per wave)
  const int row0 = m0 + ml;
  const float* __restrict__ xr0 = x   + (size_t)row0 * EDGES;
  const float* __restrict__ xr1 = x   + (size_t)(row0 + 16) * EDGES;
  const float* __restrict__ lr0 = llr + (size_t)row0 * NVARS;
  const float* __restrict__ lr1 = llr + (size_t)(row0 + 16) * NVARS;

  v8f acc[2][4];
#pragma unroll
  for (int p = 0; p < 2; p++)
#pragma unroll
    for (int t = 0; t < 4; t++) acc[p][t] = (v8f){};

  for (int c = 0; c < NCHUNK; c++) {
    const int kc = c * BK;

    __syncthreads();   // previous chunk fully consumed
    // ---- stage B chunk (swizzled) : 4 reps of 32B per thread ----
#pragma unroll
    for (int rep = 0; rep < 4; rep++) {
      // global k for this 32B run: kc + rep*32 + st_h*16
      v16h v = *(const v16h*)(wsrc + kc + rep * 32);
      *(v16h*)(ldst + rep * 256 * 16) = v;
    }
    __syncthreads();   // B chunk visible

    // ---- A source for this chunk (never straddles the 2048 boundary) ----
    const float* __restrict__ a0;
    const float* __restrict__ a1;
    if (c < XCHUNK) { a0 = xr0 + kc;          a1 = xr1 + kc; }
    else            { a0 = lr0 + (kc - EDGES); a1 = lr1 + (kc - EDGES); }
    __builtin_prefetch(a0 + BK, 0, 3);
    __builtin_prefetch(a1 + BK, 0, 3);

#pragma unroll
    for (int s = 0; s < 4; s++) {
      const int ko = s * 32 + h * 16;
      // A strip 0
      f32x4 p0 = *(const f32x4*)(a0 + ko);
      f32x4 p1 = *(const f32x4*)(a0 + ko + 4);
      f32x4 p2 = *(const f32x4*)(a0 + ko + 8);
      f32x4 p3 = *(const f32x4*)(a0 + ko + 12);
      // A strip 1
      f32x4 q0 = *(const f32x4*)(a1 + ko);
      f32x4 q1 = *(const f32x4*)(a1 + ko + 4);
      f32x4 q2 = *(const f32x4*)(a1 + ko + 8);
      f32x4 q3 = *(const f32x4*)(a1 + ko + 12);

      v16h aA, aB;
#pragma unroll
      for (int i = 0; i < 4; i++) {
        aA[i] = (_Float16)p0[i]; aA[4 + i] = (_Float16)p1[i];
        aA[8 + i] = (_Float16)p2[i]; aA[12 + i] = (_Float16)p3[i];
        aB[i] = (_Float16)q0[i]; aB[4 + i] = (_Float16)q1[i];
        aB[8 + i] = (_Float16)q2[i]; aB[12 + i] = (_Float16)q3[i];
      }

#pragma unroll
      for (int t = 0; t < 4; t++) {
        const int g = s * 8 + nw * 4 + t;
        v16h b = *(const v16h*)(&ldsB[(size_t)(g * 32 + lane) * 16]);
        acc[0][t] = __builtin_amdgcn_wmma_f32_16x16x32_f16(
            false, aA, false, b, (short)0, acc[0][t], false, false);
        acc[1][t] = __builtin_amdgcn_wmma_f32_16x16x32_f16(
            false, aB, false, b, (short)0, acc[1][t], false, false);
      }
    }
  }

  // ---- epilogue: C layout -> (M = r + 8h, N = ml) per 16x16 tile ----
#pragma unroll
  for (int p = 0; p < 2; p++) {
#pragma unroll
    for (int r = 0; r < 8; r++) {
      const int orow = m0 + p * 16 + 8 * h + r;
      float* __restrict__ op =
          out + (size_t)orow * EDGES + ncol0 + nw * 64 + ml;
#pragma unroll
      for (int t = 0; t < 4; t++) {
        float v = fminf(fmaxf(acc[p][t][r], -CLIP_TANH), CLIP_TANH);
        op[t * 16] = tanhf(0.5f * v);
      }
    }
  }
}

extern "C" void kernel_launch(void* const* d_in, const int* in_sizes, int n_in,
                              void* d_out, int out_size, void* d_ws, size_t ws_size,
                              hipStream_t stream) {
  const float* x      = (const float*)d_in[0];  // [BATCH, EDGES]
  const float* llr    = (const float*)d_in[1];  // [BATCH, NVARS]
  const float* u      = (const float*)d_in[2];  // [EDGES]
  const float* odd_w  = (const float*)d_in[3];  // [EDGES, EDGES]
  const float* llr_w  = (const float*)d_in[4];  // [NVARS, EDGES]
  const float* logits = (const float*)d_in[5];  // [EDGES]
  const float* m_o2e  = (const float*)d_in[6];  // [EDGES, EDGES]
  const float* m_skip = (const float*)d_in[7];  // [NVARS, EDGES]
  _Float16* wct = (_Float16*)d_ws;              // [EDGES, KTOT] f16, ~10 MB
  float* out = (float*)d_out;                   // [BATCH, EDGES]

  {
    size_t total = (size_t)EDGES * KTOT;        // 5,242,880
    prep_weights<<<dim3((unsigned)(total / 256)), 256, 0, stream>>>(
        u, odd_w, llr_w, logits, m_o2e, m_skip, wct);
  }

  dim3 grid(EDGES / 128, BATCH / 128);          // (16, 128)
  decoder_gemm<<<grid, 256, 0, stream>>>(x, llr, wct, out);
}
